// RNN_23287312678992
// MI455X (gfx1250) — matrix operand order repaired and checked by reading, not compiled, not run on hardware
//
#include <hip/hip_runtime.h>
#include <hip/hip_bf16.h>
#include <math.h>

// ---------------------------------------------------------------------------
// Problem constants (match reference): B=2048, T=256, I=64, H=256, C=1
// ---------------------------------------------------------------------------
#define BB 2048
#define TT 256
#define II 64
#define HH 256

typedef _Float16 h16 __attribute__((ext_vector_type(16)));
typedef _Float16 h8  __attribute__((ext_vector_type(8)));
typedef float    f8  __attribute__((ext_vector_type(8)));

// ---------------------------------------------------------------------------
// WMMA helpers (CDNA5 v_wmma_f32_16x16x32_f16, wave32)
// Fragment layouts per CDNA5 ISA §7.12.2.
// ---------------------------------------------------------------------------
__device__ __forceinline__ f8 wmma_f16(h16 a, h16 b, f8 c) {
    // (neg_a, A, neg_b, B, c_mod, C, reuse_a, reuse_b)
    return __builtin_amdgcn_wmma_f32_16x16x32_f16(false, a, false, b,
                                                  (short)0, c, false, false);
}

// A fragment: 16x32 f16 tile from row-major LDS [rows][ldk].
// lanes 0-15 : M=lane,    K = kBase+0..7  and kBase+16..23
// lanes 16-31: M=lane-16, K = kBase+8..15 and kBase+24..31
__device__ __forceinline__ h16 ldA(const _Float16* __restrict__ base,
                                   int rowBase, int kBase, int ldk) {
    const int lane = threadIdx.x & 31;
    const int r    = rowBase + (lane & 15);
    const int koff = (lane < 16) ? 0 : 8;
    const _Float16* p = base + r * ldk + kBase + koff;
    h8 lo = *(const h8*)(p);        // ds_load_b128
    h8 hi = *(const h8*)(p + 16);   // ds_load_b128
    h16 a;
#pragma unroll
    for (int i = 0; i < 8; ++i) { a[i] = lo[i]; a[8 + i] = hi[i]; }
    return a;
}

// B fragment: 32x16 tile of W^T where W is row-major LDS [N][ldk] (w[n][k]).
// lanes 0-15 : N=lane,    K = kBase+0..7  and kBase+16..23
// lanes 16-31: N=lane-16, K = kBase+8..15 and kBase+24..31
__device__ __forceinline__ h16 ldB(const _Float16* __restrict__ w,
                                   int colBase, int kBase, int ldk) {
    const int lane = threadIdx.x & 31;
    const int c    = colBase + (lane & 15);
    const int koff = (lane < 16) ? 0 : 8;
    const _Float16* p = w + c * ldk + kBase + koff;
    h8 lo = *(const h8*)(p);
    h8 hi = *(const h8*)(p + 16);
    h16 b;
#pragma unroll
    for (int i = 0; i < 8; ++i) { b[i] = lo[i]; b[8 + i] = hi[i]; }
    return b;
}

// Stage a row-major f32 matrix from global into LDS as f16 (block-cooperative).
__device__ __forceinline__ void stage_f32_to_f16(const float* __restrict__ g,
                                                 _Float16* __restrict__ s,
                                                 int elems, int tid) {
    for (int i = tid * 4; i < elems; i += 256 * 4) {
        float4 v = *(const float4*)(g + i);
        s[i + 0] = (_Float16)v.x; s[i + 1] = (_Float16)v.y;
        s[i + 2] = (_Float16)v.z; s[i + 3] = (_Float16)v.w;
    }
}

// ---------------------------------------------------------------------------
// CDNA5 async global->LDS copy (ASYNCcnt-tracked), inline asm per
// CDNA5 ISA. Each enabled lane moves 16 bytes.
// LDS destination address = low 32 bits of the generic pointer (LDS aperture).
// ---------------------------------------------------------------------------
__device__ __forceinline__ void async_copy_b128(void* ldsDst,
                                                const void* gSrc) {
    const uint32_t ldsAddr = (uint32_t)(uintptr_t)ldsDst;
    const uint64_t gAddr   = (uint64_t)(uintptr_t)gSrc;
    asm volatile("global_load_async_to_lds_b128 %0, %1, off"
                 :
                 : "v"(ldsAddr), "v"(gAddr)
                 : "memory");
}
__device__ __forceinline__ void wait_async0() {
    asm volatile("s_wait_asynccnt 0x0" ::: "memory");
}

// Issue this wave's 16x64 xp sub-tile (rows tm*16.., cols tnB*16..) for step t
// into sXP[t&1]: 4 x b128 per wave (32 lanes x 16B = 4 rows per issue).
__device__ __forceinline__ void issue_xp_tile(const _Float16* __restrict__ xp,
                                              _Float16* __restrict__ sXP,
                                              int bBase, int tm, int tnB,
                                              int t, int T, int lane) {
    _Float16* dst = sXP + (t & 1) * (32 * HH);
    const int r  = lane >> 3;        // 0..3
    const int c  = (lane & 7) * 8;   // 16B chunk within 64 cols
#pragma unroll
    for (int l = 0; l < 4; ++l) {
        const int row = tm * 16 + l * 4 + r;
        const int col = tnB * 16 + c;
        const _Float16* g =
            xp + ((size_t)(bBase + row) * T + t) * HH + col;
        async_copy_b128(dst + row * HH + col, g);
    }
}

// ---------------------------------------------------------------------------
// Kernel 1: xp0 = f16( x @ w_ih0^T + b_ih0 + b_hh0 ),  K = I = 64.
// 64 rows/block, N = 256, A (x, f32) and W staged to LDS as f16.
// ---------------------------------------------------------------------------
__global__ void __launch_bounds__(256)
gemm_xp0(const float* __restrict__ x, const float* __restrict__ W,
         const float* __restrict__ bias1, const float* __restrict__ bias2,
         _Float16* __restrict__ out) {
    extern __shared__ char smem[];
    _Float16* sW = (_Float16*)smem;          // [256][64]
    _Float16* sA = sW + 256 * II;            // [64][64]

    const int tid   = threadIdx.x;
    const int rBase = blockIdx.x * 64;

    stage_f32_to_f16(W, sW, 256 * II, tid);
    stage_f32_to_f16(x + (size_t)rBase * II, sA, 64 * II, tid);
    __syncthreads();

    const int wave = tid >> 5;
    const int lane = tid & 31;
    const int tm   = wave >> 1;              // 8 tiles/wave share one M-tile

    h16 afrag[II / 32];
#pragma unroll
    for (int kc = 0; kc < II / 32; ++kc)
        afrag[kc] = ldA(sA, tm * 16, kc * 32, II);

    for (int i = 0; i < 8; ++i) {
        const int tn = (wave & 1) * 8 + i;
        const int n  = tn * 16 + (lane & 15);
        const float bn = bias1[n] + bias2[n];
        f8 acc;
#pragma unroll
        for (int j = 0; j < 8; ++j) acc[j] = bn;
#pragma unroll
        for (int kc = 0; kc < II / 32; ++kc) {
            h16 b = ldB(sW, tn * 16, kc * 32, II);
            acc = wmma_f16(afrag[kc], b, acc);
        }
#pragma unroll
        for (int j = 0; j < 8; ++j) {
            const int m = tm * 16 + j + ((lane < 16) ? 0 : 8);
            out[(size_t)(rBase + m) * HH + n] = (_Float16)acc[j];
        }
    }
}

// ---------------------------------------------------------------------------
// Kernel 2 (fused): layer-0 recurrence + layer-1 input projection.
//   h0_t  = tanh(xp0_t + h0_{t-1} @ Whh0^T)
//   xp1_t = h0_t @ Wih1^T + b_ih1 + b_hh1          (written f16 to global)
// LDS: Whh0 (128K) + Wih1 (128K) + h double-buffer (32K) + async xp
// double-buffer (32K) = 320 KB (WGP max). ROWS = 32 batch rows per block.
// xp pipeline: per step, wait own ASYNCcnt==0, issue next step's sub-tile
// (each wave loads exactly the region it consumes -> no barrier needed),
// then compute. One __syncthreads per step for the h state handoff.
// ---------------------------------------------------------------------------
__global__ void __launch_bounds__(256)
rnn_scan0_fused(const _Float16* __restrict__ xp0,  // [B][T][H] f16 (pre-biased)
                const float* __restrict__ Whh0,    // [H][H] f32
                const float* __restrict__ Wih1,    // [H][H] f32
                const float* __restrict__ b_ih1,
                const float* __restrict__ b_hh1,
                _Float16* __restrict__ xp1,        // [B][T][H] f16 out
                int T) {
    constexpr int ROWS = 32;
    extern __shared__ char smem[];
    _Float16* sWhh = (_Float16*)smem;        // [256][256] 128 KB
    _Float16* sWih = sWhh + HH * HH;         // [256][256] 128 KB
    _Float16* hbuf = sWih + HH * HH;         // [2][ROWS][256] 32 KB
    _Float16* sXP  = hbuf + 2 * ROWS * HH;   // [2][ROWS][256] 32 KB

    const int tid   = threadIdx.x;
    const int bBase = blockIdx.x * ROWS;
    const int wave  = tid >> 5;
    const int lane  = tid & 31;
    const int tm    = (wave >> 2);           // waves 0-3 -> tm 0, 4-7 -> tm 1
    const int tnB   = (wave & 3) * 4;        // 4 N-tiles per wave

    // Prime the xp pipeline with step 0 while weights stage.
    issue_xp_tile(xp0, sXP, bBase, tm, tnB, 0, T, lane);

    stage_f32_to_f16(Whh0, sWhh, HH * HH, tid);
    stage_f32_to_f16(Wih1, sWih, HH * HH, tid);
    for (int i = tid * 8; i < 2 * ROWS * HH; i += 256 * 8) {
        uint4 z = {0, 0, 0, 0};
        *(uint4*)(hbuf + i) = z;
    }
    __syncthreads();

    int cur = 0;
    for (int t = 0; t < T; ++t) {
        const _Float16* hs = hbuf + cur * ROWS * HH;
        _Float16* hd       = hbuf + (cur ^ 1) * ROWS * HH;

        wait_async0();                        // step-t xp sub-tile resident
        if (t + 1 < T)                        // overlap t+1 copy with compute
            issue_xp_tile(xp0, sXP, bBase, tm, tnB, t + 1, T, lane);
        const _Float16* sxp = sXP + (t & 1) * (ROWS * HH);

        // ---- phase 1: h0_t = tanh(xp0_t + hs @ Whh0^T) -> hd -------------
        h16 afrag[8];
#pragma unroll
        for (int kc = 0; kc < 8; ++kc)
            afrag[kc] = ldA(hs, tm * 16, kc * 32, HH);

#pragma unroll
        for (int i = 0; i < 4; ++i) {
            const int tn = tnB + i;
            const int n  = tn * 16 + (lane & 15);
            f8 acc;
#pragma unroll
            for (int j = 0; j < 8; ++j) {
                const int m = tm * 16 + j + ((lane < 16) ? 0 : 8);
                acc[j] = (float)sxp[m * HH + n];
            }
#pragma unroll
            for (int kc = 0; kc < 8; ++kc) {
                h16 b = ldB(sWhh, tn * 16, kc * 32, HH);
                acc = wmma_f16(afrag[kc], b, acc);
            }
#pragma unroll
            for (int j = 0; j < 8; ++j) {
                const int m = tm * 16 + j + ((lane < 16) ? 0 : 8);
                hd[m * HH + n] = (_Float16)tanhf(acc[j]);
            }
        }
        __syncthreads();   // hd complete; also fences next step's hs writes

        // ---- phase 2: xp1_t = hd @ Wih1^T + bias -> global ---------------
        h16 afrag2[8];
#pragma unroll
        for (int kc = 0; kc < 8; ++kc)
            afrag2[kc] = ldA(hd, tm * 16, kc * 32, HH);

#pragma unroll
        for (int i = 0; i < 4; ++i) {
            const int tn = tnB + i;
            const int n  = tn * 16 + (lane & 15);
            const float bn = b_ih1[n] + b_hh1[n];
            f8 acc;
#pragma unroll
            for (int j = 0; j < 8; ++j) acc[j] = bn;
#pragma unroll
            for (int kc = 0; kc < 8; ++kc) {
                h16 b = ldB(sWih, tn * 16, kc * 32, HH);
                acc = wmma_f16(afrag2[kc], b, acc);
            }
#pragma unroll
            for (int j = 0; j < 8; ++j) {
                const int m = tm * 16 + j + ((lane < 16) ? 0 : 8);
                xp1[((size_t)(bBase + m) * T + t) * HH + n] = (_Float16)acc[j];
            }
        }
        cur ^= 1;
    }
}

// ---------------------------------------------------------------------------
// Kernel 3: layer-1 recurrence, emitting only the final hidden state.
//   h1_t = tanh(xp1_t + h1_{t-1} @ Whh1^T);  out = h1_{T-1}  (f16 [B][H])
// Same async xp pipeline; LDS = 128K + 32K + 32K = 192 KB.
// ---------------------------------------------------------------------------
__global__ void __launch_bounds__(256)
rnn_scan1_last(const _Float16* __restrict__ xp1,  // [B][T][H] f16 (pre-biased)
               const float* __restrict__ Whh1,    // [H][H] f32
               _Float16* __restrict__ hlast,      // [B][H] f16
               int T) {
    constexpr int ROWS = 32;
    extern __shared__ char smem[];
    _Float16* sW   = (_Float16*)smem;        // [256][256] 128 KB
    _Float16* hbuf = sW + HH * HH;           // [2][ROWS][256] 32 KB
    _Float16* sXP  = hbuf + 2 * ROWS * HH;   // [2][ROWS][256] 32 KB

    const int tid   = threadIdx.x;
    const int bBase = blockIdx.x * ROWS;
    const int wave  = tid >> 5;
    const int lane  = tid & 31;
    const int tm    = (wave >> 2);
    const int tnB   = (wave & 3) * 4;

    issue_xp_tile(xp1, sXP, bBase, tm, tnB, 0, T, lane);

    stage_f32_to_f16(Whh1, sW, HH * HH, tid);
    for (int i = tid * 8; i < 2 * ROWS * HH; i += 256 * 8) {
        uint4 z = {0, 0, 0, 0};
        *(uint4*)(hbuf + i) = z;
    }
    __syncthreads();

    int cur = 0;
    for (int t = 0; t < T; ++t) {
        const _Float16* hs = hbuf + cur * ROWS * HH;
        _Float16* hd       = hbuf + (cur ^ 1) * ROWS * HH;

        wait_async0();
        if (t + 1 < T)
            issue_xp_tile(xp1, sXP, bBase, tm, tnB, t + 1, T, lane);
        const _Float16* sxp = sXP + (t & 1) * (ROWS * HH);

        h16 afrag[8];
#pragma unroll
        for (int kc = 0; kc < 8; ++kc)
            afrag[kc] = ldA(hs, tm * 16, kc * 32, HH);

#pragma unroll
        for (int i = 0; i < 4; ++i) {
            const int tn = tnB + i;
            const int n  = tn * 16 + (lane & 15);
            f8 acc;
#pragma unroll
            for (int j = 0; j < 8; ++j) {
                const int m = tm * 16 + j + ((lane < 16) ? 0 : 8);
                acc[j] = (float)sxp[m * HH + n];
            }
#pragma unroll
            for (int kc = 0; kc < 8; ++kc) {
                h16 b = ldB(sW, tn * 16, kc * 32, HH);
                acc = wmma_f16(afrag[kc], b, acc);
            }
#pragma unroll
            for (int j = 0; j < 8; ++j) {
                const int m = tm * 16 + j + ((lane < 16) ? 0 : 8);
                const _Float16 hv = (_Float16)tanhf(acc[j]);
                hd[m * HH + n] = hv;
                if (t == T - 1)
                    hlast[(size_t)(bBase + m) * HH + n] = hv;
            }
        }
        __syncthreads();
        cur ^= 1;
    }
}

// ---------------------------------------------------------------------------
// Kernel 4: out[b] = sigmoid(dot(h2[b], w_fc) + b_fc). One wave per row.
// ---------------------------------------------------------------------------
__global__ void __launch_bounds__(256)
fc_sigmoid(const _Float16* __restrict__ h2, const float* __restrict__ wfc,
           const float* __restrict__ bfc, float* __restrict__ out, int Btot) {
    const int wave = threadIdx.x >> 5;
    const int lane = threadIdx.x & 31;
    const int row  = blockIdx.x * 8 + wave;
    if (row >= Btot) return;
    float s = 0.f;
#pragma unroll
    for (int k = lane; k < HH; k += 32)
        s += (float)h2[(size_t)row * HH + k] * wfc[k];
#pragma unroll
    for (int off = 16; off; off >>= 1) s += __shfl_xor(s, off, 32);
    if (lane == 0) out[row] = 1.f / (1.f + expf(-(s + bfc[0])));
}

// ---------------------------------------------------------------------------
// Launch
// ---------------------------------------------------------------------------
extern "C" void kernel_launch(void* const* d_in, const int* in_sizes, int n_in,
                              void* d_out, int out_size, void* d_ws,
                              size_t ws_size, hipStream_t stream) {
    const float* x     = (const float*)d_in[0];   // (B,T,I)
    const float* w_ih0 = (const float*)d_in[1];   // (H,I)
    const float* w_hh0 = (const float*)d_in[2];   // (H,H)
    const float* b_ih0 = (const float*)d_in[3];
    const float* b_hh0 = (const float*)d_in[4];
    const float* w_ih1 = (const float*)d_in[5];   // (H,H)
    const float* w_hh1 = (const float*)d_in[6];   // (H,H)
    const float* b_ih1 = (const float*)d_in[7];
    const float* b_hh1 = (const float*)d_in[8];
    const float* w_fc  = (const float*)d_in[9];   // (1,H)
    const float* b_fc  = (const float*)d_in[10];

    const size_t seqElems = (size_t)BB * TT * HH;     // 134M
    _Float16* xp0 = (_Float16*)d_ws;                  // [B][T][H] f16
    _Float16* xp1 = xp0 + seqElems;                   // [B][T][H] f16
    _Float16* h2l = xp1 + seqElems;                   // [B][H]    f16

    // 1) xp0 = f16(x @ w_ih0^T + b_ih0 + b_hh0)
    {
        const size_t lds = (size_t)(256 * II + 64 * II) * sizeof(_Float16);
        gemm_xp0<<<(BB * TT) / 64, 256, lds, stream>>>(x, w_ih0, b_ih0, b_hh0,
                                                       xp0);
    }
    // 2) fused layer-0 recurrence + layer-1 input projection -> xp1
    {
        const size_t lds =
            (size_t)(2 * HH * HH + 4 * 32 * HH) * sizeof(_Float16);  // 320 KB
        rnn_scan0_fused<<<BB / 32, 256, lds, stream>>>(xp0, w_hh0, w_ih1,
                                                       b_ih1, b_hh1, xp1, TT);
    }
    // 3) layer-1 recurrence -> last hidden
    {
        const size_t lds =
            (size_t)(HH * HH + 4 * 32 * HH) * sizeof(_Float16);      // 192 KB
        rnn_scan1_last<<<BB / 32, 256, lds, stream>>>(xp1, w_hh1, h2l, TT);
    }
    // 4) sigmoid head
    fc_sigmoid<<<BB / 8, 256, 0, stream>>>(h2l, w_fc, b_fc, (float*)d_out, BB);
}
